// SinkhornOT_61134564491497
// MI455X (gfx1250) — matrix operand order repaired and checked by reading, not compiled, not run on hardware
//
#include <hip/hip_runtime.h>
#include <math.h>

// ---------------------------------------------------------------------------
// Sinkhorn OT for MI455X (gfx1250, wave32, WMMA + async-to-LDS staging).
//
// n = m = 8192, d = 1024, REG = 0.1, 100 iterations.
// K (Gibbs kernel) kept in bf16 (128 MB) so it fits the 192 MB L2: all 200
// Sinkhorn matvec passes stream from L2. Both big GEMMs use
// v_wmma_f32_16x16x32_bf16 with 128x128 block tiles; A/B k-slices are staged
// into double-buffered LDS via GLOBAL_LOAD_ASYNC_TO_LDS_B128 (ASYNCcnt) when
// the toolchain exposes the builtin, else a synchronous global->LDS copy.
// ---------------------------------------------------------------------------

typedef __attribute__((ext_vector_type(16))) __bf16   v16bf;
typedef __attribute__((ext_vector_type(8)))  float    v8f;
typedef __attribute__((ext_vector_type(4)))  unsigned int u32x4;
typedef int i32v4 __attribute__((vector_size(16)));   // GCC-vector to match builtin

#define N_ROWS 8192
#define N_DIM  1024
#define INV_N  (1.0f / 8192.0f)

#define GLOBAL_AS __attribute__((address_space(1)))
#define LDS_AS    __attribute__((address_space(3)))

#if defined(__has_builtin)
# if __has_builtin(__builtin_amdgcn_global_load_async_to_lds_b128)
#  define USE_ASYNC_LDS 1
# endif
#endif

static __device__ __forceinline__ void async_copy_b128(const __bf16* g, __bf16* l) {
#if defined(USE_ASYNC_LDS)
  __builtin_amdgcn_global_load_async_to_lds_b128((GLOBAL_AS i32v4*)g,
                                                 (LDS_AS i32v4*)l, 0, 0);
#else
  *(u32x4*)l = *(const u32x4*)g;
#endif
}

static __device__ __forceinline__ void wait_async_lds() {
#if defined(USE_ASYNC_LDS)
# if __has_builtin(__builtin_amdgcn_s_wait_asynccnt)
  __builtin_amdgcn_s_wait_asynccnt(0);
# elif defined(__AMDGCN__)
  asm volatile("s_wait_asynccnt 0" ::: "memory");
# endif
#endif
}

static __device__ __forceinline__ v8f wmma_bf16(v16bf a, v16bf b, v8f c) {
  // (neg_a, A, neg_b, B, c_mod, C, reuse_a, reuse_b)
  return __builtin_amdgcn_wmma_f32_16x16x32_bf16(false, a, false, b, (short)0, c,
                                                 false, false);
}

static __device__ __forceinline__ v16bf load_frag(const __bf16* p0, const __bf16* p1) {
  union { v16bf v; u32x4 q[2]; } f;
  f.q[0] = *reinterpret_cast<const u32x4*>(p0);
  f.q[1] = *reinterpret_cast<const u32x4*>(p1);
  return f.v;
}

static __device__ __forceinline__ float blockReduceSum256(float x) {
  __shared__ float red[256];
  const int t = threadIdx.x;
  red[t] = x;
  __syncthreads();
  for (int s = 128; s > 0; s >>= 1) {
    if (t < s) red[t] += red[t + s];
    __syncthreads();
  }
  return red[0];
}

// --- 1) L2-normalize rows of X and Y, write bf16 ---------------------------
__global__ __launch_bounds__(256) void k_rownorm(const float* __restrict__ X,
                                                 const float* __restrict__ Y,
                                                 __bf16* __restrict__ Xn,
                                                 __bf16* __restrict__ Yn) {
  const int row = blockIdx.x;               // 0..16383
  const float* src = (row < N_ROWS) ? (X + (size_t)row * N_DIM)
                                    : (Y + (size_t)(row - N_ROWS) * N_DIM);
  __bf16* dst = (row < N_ROWS) ? (Xn + (size_t)row * N_DIM)
                               : (Yn + (size_t)(row - N_ROWS) * N_DIM);
  float ss = 0.f;
  float vals[4];
#pragma unroll
  for (int k = 0; k < 4; ++k) {
    vals[k] = src[threadIdx.x + k * 256];
    ss += vals[k] * vals[k];
  }
  const float tot = blockReduceSum256(ss);
  const float scale = 1.0f / fmaxf(sqrtf(tot), 1e-12f);
#pragma unroll
  for (int k = 0; k < 4; ++k)
    dst[threadIdx.x + k * 256] = (__bf16)(vals[k] * scale);
}

// --- 2) Transpose Yn (8192 x 1024) -> YnT (1024 x 8192), bf16, LDS tiled ---
__global__ __launch_bounds__(256) void k_transpose(const __bf16* __restrict__ Yn,
                                                   __bf16* __restrict__ YnT) {
  __shared__ __bf16 tile[32][33];
  const int tx = threadIdx.x, ty = threadIdx.y;     // (32, 8)
  const int j0 = blockIdx.x * 32;                   // row block in Yn
  const int d0 = blockIdx.y * 32;                   // col block in Yn
#pragma unroll
  for (int i = 0; i < 32; i += 8)
    tile[ty + i][tx] = Yn[(size_t)(j0 + ty + i) * N_DIM + d0 + tx];
  __syncthreads();
#pragma unroll
  for (int i = 0; i < 32; i += 8)
    YnT[(size_t)(d0 + ty + i) * N_ROWS + j0 + tx] = tile[tx][ty + i];
}

// --- 3/7) 128x128-tile WMMA GEMM with double-buffered LDS staging ----------
// A: row-major (lda = KMAX). B: row-major (ldb = KMAX); computes A * B^T.
// EXP_EPI: D = bf16 exp(10*(s-1)) (Gibbs kernel build), else D = f32 s.
#define LDS_STRIDE 40    // 32 k-elements + 8 pad (80B rows spread LDS banks)

template <int KMAX, int LDD, bool EXP_EPI, typename DT>
__global__ __launch_bounds__(256) void k_gemm_lds(const __bf16* __restrict__ A,
                                                  const __bf16* __restrict__ B,
                                                  DT* __restrict__ D) {
  __shared__ __bf16 sA[2][128 * LDS_STRIDE];
  __shared__ __bf16 sB[2][128 * LDS_STRIDE];

  const int lane = threadIdx.x & 31;
  const int wave = threadIdx.x >> 5;                // 0..7
  const int wr = wave >> 1;                         // 0..3 -> 32-row strip
  const int wc = wave & 1;                          // 0..1 -> 64-col strip
  const int rowBlk = blockIdx.y * 128;
  const int colBlk = blockIdx.x * 128;
  const int mlane = lane & 15;
  const int half  = lane >> 4;

  // Stage one 128x32 k-slice of A and B into LDS buffer `buf`.
  auto stage = [&](int k0, int buf) {
#pragma unroll
    for (int i = 0; i < 2; ++i) {
      const int c   = threadIdx.x + 256 * i;        // 0..511 chunk id
      const int row = c >> 2;                       // 128 rows
      const int ch  = c & 3;                        // 4 x 8-elem chunks
      async_copy_b128(A + (size_t)(rowBlk + row) * KMAX + k0 + ch * 8,
                      &sA[buf][row * LDS_STRIDE + ch * 8]);
      async_copy_b128(B + (size_t)(colBlk + row) * KMAX + k0 + ch * 8,
                      &sB[buf][row * LDS_STRIDE + ch * 8]);
    }
  };

  v8f acc[2][4] = {};
  stage(0, 0);
  for (int k0 = 0; k0 < KMAX; k0 += 32) {
    const int buf = (k0 >> 5) & 1;
    wait_async_lds();
    __syncthreads();                 // tile `buf` ready; prev compute done
    if (k0 + 32 < KMAX) stage(k0 + 32, buf ^ 1);

    v16bf a[2], b[4];
#pragma unroll
    for (int mt = 0; mt < 2; ++mt) {
      const __bf16* p = &sA[buf][(wr * 32 + mt * 16 + mlane) * LDS_STRIDE + half * 8];
      a[mt] = load_frag(p, p + 16);
    }
#pragma unroll
    for (int nt = 0; nt < 4; ++nt) {
      const __bf16* p = &sB[buf][(wc * 64 + nt * 16 + mlane) * LDS_STRIDE + half * 8];
      b[nt] = load_frag(p, p + 16);
    }
#pragma unroll
    for (int mt = 0; mt < 2; ++mt)
#pragma unroll
      for (int nt = 0; nt < 4; ++nt)
        acc[mt][nt] = wmma_bf16(a[mt], b[nt], acc[mt][nt]);
  }

  // Epilogue: C layout — VGPR r: lanes0-15 -> M=r, lanes16-31 -> M=8+r.
#pragma unroll
  for (int mt = 0; mt < 2; ++mt)
#pragma unroll
    for (int nt = 0; nt < 4; ++nt)
#pragma unroll
      for (int r = 0; r < 8; ++r) {
        const int row = rowBlk + wr * 32 + mt * 16 + r + 8 * half;
        const int col = colBlk + wc * 64 + nt * 16 + mlane;
        const float s = acc[mt][nt][r];
        if (EXP_EPI)
          D[(size_t)row * LDD + col] = (DT)__expf(10.0f * (s - 1.0f));
        else
          D[(size_t)row * LDD + col] = (DT)s;
      }
}

// --- 4) u = v = 1/n --------------------------------------------------------
__global__ void k_init_uv(float* __restrict__ u, float* __restrict__ v) {
  const int i = blockIdx.x * 256 + threadIdx.x;
  u[i] = INV_N;
  v[i] = INV_N;
}

// --- 5a) partial[rb][j] = sum over 1024 rows of K[i][j]*u[i] ---------------
__global__ __launch_bounds__(256) void k_ktu_partial(const __bf16* __restrict__ Km,
                                                     const float* __restrict__ u,
                                                     float* __restrict__ partial) {
  const int j  = blockIdx.x * 256 + threadIdx.x;    // column
  const int r0 = blockIdx.y * 1024;                 // row chunk
  __shared__ float su[256];
  float acc = 0.f;
  for (int ib = 0; ib < 1024; ib += 256) {
    __syncthreads();
    su[threadIdx.x] = u[r0 + ib + threadIdx.x];
    __syncthreads();
#pragma unroll 4
    for (int t = 0; t < 256; ++t)
      acc += (float)Km[(size_t)(r0 + ib + t) * N_ROWS + j] * su[t];
  }
  partial[(size_t)blockIdx.y * N_ROWS + j] = acc;
}

// --- 5b) v[j] = b / sum_rb partial[rb][j] ----------------------------------
__global__ void k_v_update(const float* __restrict__ partial, float* __restrict__ v) {
  const int j = blockIdx.x * 256 + threadIdx.x;
  float s = 0.f;
#pragma unroll
  for (int rb = 0; rb < 8; ++rb) s += partial[(size_t)rb * N_ROWS + j];
  v[j] = INV_N / s;
}

// --- 5c) u[i] = a / (K v)_i  (one block per row, LDS reduce) ---------------
__global__ __launch_bounds__(256) void k_kv_u(const __bf16* __restrict__ Km,
                                              const float* __restrict__ v,
                                              float* __restrict__ u) {
  const size_t row = blockIdx.x;
  float acc = 0.f;
#pragma unroll 4
  for (int j = threadIdx.x; j < N_ROWS; j += 256)
    acc += (float)Km[row * N_ROWS + j] * v[j];
  const float tot = blockReduceSum256(acc);
  if (threadIdx.x == 0) u[row] = INV_N / tot;
}

// --- 6) YnT[d][j] *= v[j]  (in place; YnT rebuilt every launch) ------------
__global__ void k_scale_v(__bf16* __restrict__ YnT, const float* __restrict__ v) {
  const size_t idx = (size_t)blockIdx.x * 256 + threadIdx.x;  // over 1024*8192
  const int j = (int)(idx & (N_ROWS - 1));
  YnT[idx] = (__bf16)((float)YnT[idx] * v[j]);
}

// --- 8) l2-normalize rows of out in place ----------------------------------
__global__ __launch_bounds__(256) void k_norm_out(float* __restrict__ out) {
  float* row = out + (size_t)blockIdx.x * N_DIM;
  float ss = 0.f;
  float vals[4];
#pragma unroll
  for (int k = 0; k < 4; ++k) {
    vals[k] = row[threadIdx.x + k * 256];
    ss += vals[k] * vals[k];
  }
  const float tot = blockReduceSum256(ss);
  const float scale = 1.0f / fmaxf(sqrtf(tot), 1e-12f);
#pragma unroll
  for (int k = 0; k < 4; ++k)
    row[threadIdx.x + k * 256] = vals[k] * scale;
}

// ---------------------------------------------------------------------------
extern "C" void kernel_launch(void* const* d_in, const int* in_sizes, int n_in,
                              void* d_out, int out_size, void* d_ws, size_t ws_size,
                              hipStream_t stream) {
  (void)in_sizes; (void)n_in; (void)out_size; (void)ws_size;
  const float* X = (const float*)d_in[0];
  const float* Y = (const float*)d_in[1];
  float* out = (float*)d_out;                       // 8192 x 1024 f32

  constexpr size_t MB = 1024ull * 1024ull;
  char* ws = (char*)d_ws;
  __bf16* Xn  = (__bf16*)(ws + 0);                  // 16 MB
  __bf16* Yn  = (__bf16*)(ws + 16 * MB);            // 16 MB
  __bf16* YnT = (__bf16*)(ws + 32 * MB);            // 16 MB (later v .* YnT)
  __bf16* Km  = (__bf16*)(ws + 48 * MB);            // 128 MB (fits 192MB L2)
  float*  u   = (float*)(ws + 176 * MB);            // 32 KB
  float*  v   = (float*)(ws + 176 * MB + 32 * 1024);
  float*  part= (float*)(ws + 176 * MB + 64 * 1024);// 8*8192*4 = 256 KB

  k_rownorm<<<2 * N_ROWS, 256, 0, stream>>>(X, Y, Xn, Yn);
  k_transpose<<<dim3(N_ROWS / 32, N_DIM / 32), dim3(32, 8), 0, stream>>>(Yn, YnT);
  // K = exp(10*(Xn Yn^T - 1)) in bf16
  k_gemm_lds<N_DIM, N_ROWS, true, __bf16>
      <<<dim3(N_ROWS / 128, N_ROWS / 128), 256, 0, stream>>>(Xn, Yn, Km);
  k_init_uv<<<N_ROWS / 256, 256, 0, stream>>>(u, v);

  for (int it = 0; it < 100; ++it) {
    k_ktu_partial<<<dim3(N_ROWS / 256, 8), 256, 0, stream>>>(Km, u, part);
    k_v_update<<<N_ROWS / 256, 256, 0, stream>>>(part, v);
    k_kv_u<<<N_ROWS, 256, 0, stream>>>(Km, v, u);
  }

  k_scale_v<<<(N_DIM * N_ROWS) / 256, 256, 0, stream>>>(YnT, v);
  // out = K @ (v .* Yn)   (u row-scaling cancels in the final l2norm)
  k_gemm_lds<N_ROWS, N_DIM, false, float>
      <<<dim3(N_DIM / 128, N_ROWS / 128), 256, 0, stream>>>(Km, YnT, out);
  k_norm_out<<<N_ROWS, 256, 0, stream>>>(out);
}
